// Head_35459249996700
// MI455X (gfx1250) — matrix operand order repaired and checked by reading, not compiled, not run on hardware
//
#include <hip/hip_runtime.h>
#include <hip/hip_bf16.h>

// ---------------------------------------------------------------------------
// MI455X (gfx1250) implementation. All matmuls use v_wmma_f32_16x16x32_bf16.
// q tiles are staged to LDS via global_load_async_to_lds_b128 (ASYNCcnt path);
// k/v/mem_k tiles stream from L2 with global_prefetch hints.
// ---------------------------------------------------------------------------

typedef __bf16 bf16;
typedef __attribute__((ext_vector_type(16))) __bf16 bf16x16;
typedef __attribute__((ext_vector_type(8)))  float  f32x8;

#define WMMA_BF16(A, B, C) \
  __builtin_amdgcn_wmma_f32_16x16x32_bf16(false, (A), false, (B), (short)0, (C), false, false)

#if __has_builtin(__builtin_amdgcn_global_load_async_to_lds_b128)
#define HAVE_ASYNC_LDS 1
typedef int v4i __attribute__((ext_vector_type(4)));
typedef __attribute__((address_space(1))) v4i* gv4i_p;   // global (prints as __device__)
typedef __attribute__((address_space(3))) v4i* lv4i_p;   // LDS    (prints as __shared__)
#endif

// Stage 16 bytes per thread from global to LDS (256 threads -> 4KB tile).
__device__ __forceinline__ void stage16(const void* gsrc, void* ldst) {
#ifdef HAVE_ASYNC_LDS
  __builtin_amdgcn_global_load_async_to_lds_b128((gv4i_p)(void*)gsrc, (lv4i_p)ldst, 0, 0);
#else
  *(float4*)ldst = *(const float4*)gsrc;
#endif
}

__device__ __forceinline__ void stage_wait() {
#ifdef HAVE_ASYNC_LDS
#if __has_builtin(__builtin_amdgcn_s_wait_asynccnt)
  __builtin_amdgcn_s_wait_asynccnt(0);
#else
  asm volatile("s_wait_asynccnt 0" ::: "memory");
#endif
#endif
}

static constexpr int Bn = 4, TQ = 2048, TK = 4096, En = 1024, Dn = 128, Mn = 16384;

// workspace byte offsets (all 256B aligned)
static constexpr size_t OQ  = 0;                              // bf16 q        [8192][128]
static constexpr size_t OKF = OQ  + (size_t)8192 * 128 * 2;   // bf16 k_full   [4][4096][128]
static constexpr size_t OVT = OKF + (size_t)4 * 4096 * 128 * 2; // bf16 v_T    [4][128][4096]
static constexpr size_t OWT = OVT + (size_t)4 * 128 * 4096 * 2; // bf16 Wt     [3][128][1024]
static constexpr size_t OMK = OWT + (size_t)3 * 128 * 1024 * 2; // bf16 mem_k  [4][16384][128]
static constexpr size_t OSO = OMK + (size_t)4 * 16384 * 128 * 2; // f32 sdpa_out [8192][128]
static constexpr size_t OTS = OSO + (size_t)8192 * 128 * 4;   // f32 topk score [8192][32]
static constexpr size_t OTI = OTS + (size_t)8192 * 32 * 4;    // i32 topk idx   [8192][32]

// ---------------- conversion kernels ----------------

__global__ void conv_weights(const float* __restrict__ Wq, const float* __restrict__ Wk,
                             const float* __restrict__ Wv, bf16* __restrict__ Wt) {
  int i = blockIdx.x * blockDim.x + threadIdx.x;
  const int tot = 3 * En * Dn;
  if (i >= tot) return;
  int sel = i / (En * Dn);
  int rem = i - sel * (En * Dn);
  int d = rem / En;
  int e = rem - d * En;
  const float* W = (sel == 0) ? Wq : (sel == 1) ? Wk : Wv;
  Wt[i] = (bf16)W[(size_t)e * Dn + d];   // Wt[sel][d][e] = W[e][d]
}

__global__ void conv_prefix(const float* __restrict__ ki, const float* __restrict__ vi,
                            const int* __restrict__ idxp, bf16* __restrict__ kf,
                            bf16* __restrict__ vT) {
  const int s = idxp[0] * Dn;
  int i = blockIdx.x * blockDim.x + threadIdx.x;
  if (i >= Bn * TQ * Dn) return;
  int d = i & 127;
  int rest = i >> 7;
  int t = rest & 2047;
  int b = rest >> 11;
  kf[((size_t)b * TK + t) * Dn + d] = (bf16)ki[((size_t)b * TQ + t) * En + s + d];
  vT[((size_t)b * Dn + d) * TK + t] = (bf16)vi[((size_t)b * TQ + t) * En + s + d];
}

__global__ void conv_memk(const float* __restrict__ mkf, bf16* __restrict__ mk) {
  size_t i = (size_t)blockIdx.x * blockDim.x + threadIdx.x;
  if (i >= (size_t)Bn * Mn * Dn) return;
  mk[i] = (bf16)mkf[i];
}

__global__ void zero_f32(float* __restrict__ p, int nel) {
  int i = blockIdx.x * blockDim.x + threadIdx.x;
  if (i < nel) p[i] = 0.f;
}

// ---------------- projection GEMM: q/k/v = x @ W + b ----------------
// grid (512, 3), block 256. Each wave: one 16x16 output tile, K=1024.

__global__ __launch_bounds__(256) void proj_kernel(
    const float* __restrict__ x, const bf16* __restrict__ Wt,
    const float* __restrict__ bq, const float* __restrict__ bk, const float* __restrict__ bvp,
    bf16* __restrict__ qbf, bf16* __restrict__ kf, bf16* __restrict__ vT) {
  const int sel  = blockIdx.y;
  const int row0 = blockIdx.x * 16;
  const int lane = threadIdx.x & 31, w = threadIdx.x >> 5;
  const int h = lane >> 4, n = lane & 15;
  const float* xrow = x + (size_t)(row0 + n) * En;                  // A: lane = M row
  const bf16* wtrow = Wt + ((size_t)sel * Dn + (w * 16 + n)) * En;  // B: lane = N col
  f32x8 acc = {};
  for (int e0 = 0; e0 < En; e0 += 32) {
    bf16x16 a;
    const float* pa0 = xrow + e0 + h * 16;
    const float* pa1 = xrow + e0 + h * 16 + 8;
#pragma unroll
    for (int i = 0; i < 8; ++i) { a[i] = (bf16)pa0[i]; a[8 + i] = (bf16)pa1[i]; }
    bf16x16 b = *(const bf16x16*)(wtrow + e0 + h * 16);
    acc = WMMA_BF16(a, b, acc);
  }
  const int col = w * 16 + n;
  const float* bias = (sel == 0) ? bq : (sel == 1) ? bk : bvp;
  const float bv = bias[col];
#pragma unroll
  for (int r = 0; r < 8; ++r) {
    const int row = row0 + h * 8 + r;
    const float val = acc[r] + bv;
    const int bb = row >> 11, t = row & 2047;
    if (sel == 0)      qbf[(size_t)row * Dn + col] = (bf16)val;
    else if (sel == 1) kf[((size_t)bb * TK + 2048 + t) * Dn + col] = (bf16)val;
    else               vT[((size_t)bb * Dn + col) * TK + 2048 + t] = (bf16)val;
  }
}

// ---------------- causal flash SDPA over prefix+current ----------------
// grid (128, 4), block 256 (8 waves). Wave w handles keys {c*256+w*32 .. +31}.

__global__ __launch_bounds__(256) void sdpa_kernel(
    const bf16* __restrict__ qbf, const bf16* __restrict__ kf,
    const bf16* __restrict__ vT, float* __restrict__ so) {
  __shared__ __align__(16) bf16 qsh[16][128];  // 4KB q tile, async-staged
  __shared__ bf16 Pst[8][16][32];     // per-wave P staging (C-layout -> A-layout)
  __shared__ float sh_m[8][16], sh_l[8][16];
  const int bb = blockIdx.y, t0 = blockIdx.x * 16;
  const int tid = threadIdx.x;
  const int lane = tid & 31, w = tid >> 5;
  const int h = lane >> 4, n = lane & 15;

  // async-stage the q tile (contiguous 4KB in global): 16B per thread
  {
    const char* gsrc = (const char*)(qbf + ((size_t)bb * TQ + t0) * Dn) + tid * 16;
    char* ldst = (char*)&qsh[0][0] + tid * 16;
    stage16(gsrc, ldst);
  }
  stage_wait();
  __syncthreads();

  // preload q A-fragments (16 rows x 128 K) from LDS
  const bf16* qrow = &qsh[n][0];
  bf16x16 aq[4];
#pragma unroll
  for (int kc = 0; kc < 4; ++kc) {
    const bf16* p0 = qrow + kc * 32 + h * 16;
    const bf16* p1 = qrow + kc * 32 + h * 16 + 8;
#pragma unroll
    for (int i = 0; i < 8; ++i) { aq[kc][i] = p0[i]; aq[kc][8 + i] = p1[i]; }
  }

  f32x8 accO[8];
#pragma unroll
  for (int nt = 0; nt < 8; ++nt) accO[nt] = (f32x8){};
  float mrow[8], lrow[8];
#pragma unroll
  for (int r = 0; r < 8; ++r) { mrow[r] = -1e30f; lrow[r] = 0.f; }

  const float scale = 0.08838834764831845f;  // 1/sqrt(128)
  const int kmax = t0 + 15 + 2048;           // largest key any row here may see

  for (int c = 0; c < 16; ++c) {
    const int j0 = c * 256 + w * 32;
    if (j0 > kmax) break;
    // S = q @ k^T, two 16-key tiles
    f32x8 s0 = {}, s1 = {};
    const bf16* kr0 = kf + ((size_t)bb * TK + j0 + n) * Dn;
    const bf16* kr1 = kf + ((size_t)bb * TK + j0 + 16 + n) * Dn;
    if (j0 + 256 <= kmax) {                    // hint next chunk into cache
      __builtin_prefetch(kr0 + 256 * Dn, 0, 0);
      __builtin_prefetch(kr1 + 256 * Dn, 0, 0);
    }
#pragma unroll
    for (int kc = 0; kc < 4; ++kc) {
      bf16x16 b0 = *(const bf16x16*)(kr0 + kc * 32 + h * 16);
      s0 = WMMA_BF16(aq[kc], b0, s0);
    }
#pragma unroll
    for (int kc = 0; kc < 4; ++kc) {
      bf16x16 b1 = *(const bf16x16*)(kr1 + kc * 32 + h * 16);
      s1 = WMMA_BF16(aq[kc], b1, s1);
    }
    // masked online softmax per row (cols live across 16 lanes of each half)
#pragma unroll
    for (int r = 0; r < 8; ++r) {
      const int tcap = t0 + h * 8 + r + 2048;  // causal: key <= query + (Tk-Tq)
      float v0 = ((j0 + n) <= tcap) ? s0[r] * scale : -1e30f;
      float v1 = ((j0 + 16 + n) <= tcap) ? s1[r] * scale : -1e30f;
      float mx = fmaxf(v0, v1);
#pragma unroll
      for (int off = 8; off >= 1; off >>= 1) mx = fmaxf(mx, __shfl_xor(mx, off, 16));
      const float mnew = fmaxf(mrow[r], mx);
      const float corr = __expf(mrow[r] - mnew);
      const float e0v = __expf(v0 - mnew);
      const float e1v = __expf(v1 - mnew);
      float sum = e0v + e1v;
#pragma unroll
      for (int off = 8; off >= 1; off >>= 1) sum += __shfl_xor(sum, off, 16);
      lrow[r] = lrow[r] * corr + sum;
      mrow[r] = mnew;
#pragma unroll
      for (int nt = 0; nt < 8; ++nt) accO[nt][r] *= corr;
      Pst[w][h * 8 + r][n] = (bf16)e0v;
      Pst[w][h * 8 + r][16 + n] = (bf16)e1v;
    }
    // reload P in A-fragment layout (wave-local LDS, dscnt ordered by compiler)
    bf16x16 ap;
    {
      const bf16* pr = &Pst[w][n][0];
#pragma unroll
      for (int i = 0; i < 8; ++i) { ap[i] = pr[h * 16 + i]; ap[8 + i] = pr[h * 16 + 8 + i]; }
    }
    // O += P @ V (V transposed so B-fragments load 32B contiguous per lane)
#pragma unroll
    for (int nt = 0; nt < 8; ++nt) {
      const bf16* vr = vT + ((size_t)bb * Dn + nt * 16 + n) * TK + j0 + h * 16;
      bf16x16 bv = *(const bf16x16*)vr;
      accO[nt] = WMMA_BF16(ap, bv, accO[nt]);
    }
  }

  // cross-wave split-K flash merge via LDS stats + f32 atomics
  if (n == 0) {
#pragma unroll
    for (int r = 0; r < 8; ++r) { sh_m[w][h * 8 + r] = mrow[r]; sh_l[w][h * 8 + r] = lrow[r]; }
  }
  __syncthreads();
#pragma unroll
  for (int r = 0; r < 8; ++r) {
    const int row = h * 8 + r;
    float mt = -1e30f;
    for (int w2 = 0; w2 < 8; ++w2) mt = fmaxf(mt, sh_m[w2][row]);
    float denom = 0.f;
    for (int w2 = 0; w2 < 8; ++w2) denom += __expf(sh_m[w2][row] - mt) * sh_l[w2][row];
    const float myscale = __expf(mrow[r] - mt) / denom;
    float* orow = so + ((size_t)bb * TQ + t0 + row) * Dn;
#pragma unroll
    for (int nt = 0; nt < 8; ++nt)
      atomicAdd(orow + nt * 16 + n, accO[nt][r] * myscale);
  }
}

// ---------------- kNN memory: WMMA scores + exact top-32 ----------------
// grid (128, 4), block 256. 128 candidate scores / query / iteration.

__global__ __launch_bounds__(256) void topk_kernel(
    const bf16* __restrict__ qbf, const bf16* __restrict__ mk,
    float* __restrict__ ts, int* __restrict__ ti) {
  __shared__ __align__(16) bf16 qsh[16][128];  // 4KB q tile, async-staged
  __shared__ float ss[16][132];      // padded to dodge bank conflicts
  __shared__ float tks[16][32];
  __shared__ int   tki[16][32];
  const int bb = blockIdx.y, t0 = blockIdx.x * 16;
  const int tid = threadIdx.x;
  const int lane = tid & 31, w = tid >> 5, h = lane >> 4, n = lane & 15;

  // async-stage q tile while initializing the top-k lists
  {
    const char* gsrc = (const char*)(qbf + ((size_t)bb * TQ + t0) * Dn) + tid * 16;
    char* ldst = (char*)&qsh[0][0] + tid * 16;
    stage16(gsrc, ldst);
  }
  for (int i = tid; i < 16 * 32; i += 256) { tks[i >> 5][i & 31] = -1e30f; tki[i >> 5][i & 31] = 0; }
  stage_wait();
  __syncthreads();

  const bf16* qrow = &qsh[n][0];
  bf16x16 aq[4];
#pragma unroll
  for (int kc = 0; kc < 4; ++kc) {
    const bf16* p0 = qrow + kc * 32 + h * 16;
    const bf16* p1 = qrow + kc * 32 + h * 16 + 8;
#pragma unroll
    for (int i = 0; i < 8; ++i) { aq[kc][i] = p0[i]; aq[kc][8 + i] = p1[i]; }
  }

  float minval = -1e30f; int minpos = 0;   // live only in threads 0..15
  for (int c = 0; c < Mn / 128; ++c) {
    const int j0 = c * 128 + w * 16;
    f32x8 acc = {};
    const bf16* krow = mk + ((size_t)bb * Mn + j0 + n) * Dn;
    if (c + 1 < Mn / 128) __builtin_prefetch(krow + 128 * Dn, 0, 0);
#pragma unroll
    for (int kc = 0; kc < 4; ++kc) {
      bf16x16 b = *(const bf16x16*)(krow + kc * 32 + h * 16);
      acc = WMMA_BF16(aq[kc], b, acc);
    }
#pragma unroll
    for (int r = 0; r < 8; ++r) ss[h * 8 + r][w * 16 + n] = acc[r];
    __syncthreads();
    if (tid < 16) {
      const int row = tid;
      for (int cand = 0; cand < 128; ++cand) {
        const float s = ss[row][cand];
        if (s > minval) {                 // threshold test: most candidates fail fast
          tks[row][minpos] = s;
          tki[row][minpos] = c * 128 + cand;
          float mv = tks[row][0]; int mp = 0;
          for (int k2 = 1; k2 < 32; ++k2) {
            const float v = tks[row][k2];
            if (v < mv) { mv = v; mp = k2; }
          }
          minval = mv; minpos = mp;
        }
      }
    }
    __syncthreads();
  }
  for (int i = tid; i < 16 * 32; i += 256) {
    const int row = i >> 5, k = i & 31;
    const size_t qg = (size_t)bb * TQ + t0 + row;
    ts[qg * 32 + k] = tks[row][k];
    ti[qg * 32 + k] = tki[row][k];
  }
}

// ---------------- gathered memory attention + gate combine ----------------

__global__ __launch_bounds__(128) void memattn_kernel(
    const float* __restrict__ ts, const int* __restrict__ ti,
    const float* __restrict__ mem_vals, const float* __restrict__ so,
    const float* __restrict__ gate, float* __restrict__ out) {
  __shared__ float shw[32];
  __shared__ int   shidx[32];
  __shared__ float shsum;
  const int qg = blockIdx.x;
  const int bb = qg >> 11;
  const int tid = threadIdx.x;
  if (tid < 32) {
    float s = ts[(size_t)qg * 32 + tid] * 0.03125f;  // 1/sqrt(E)=1/32
    float mx = s;
#pragma unroll
    for (int off = 16; off >= 1; off >>= 1) mx = fmaxf(mx, __shfl_xor(mx, off, 32));
    const float e = __expf(s - mx);
    float sm = e;
#pragma unroll
    for (int off = 16; off >= 1; off >>= 1) sm += __shfl_xor(sm, off, 32);
    shw[tid] = e;
    shidx[tid] = ti[(size_t)qg * 32 + tid];
    if (tid == 0) shsum = sm;
  }
  __syncthreads();
  const float inv = 1.0f / shsum;
  float acc = 0.f;
#pragma unroll 4
  for (int k = 0; k < 32; ++k)
    acc += shw[k] * mem_vals[((size_t)bb * Mn + shidx[k]) * Dn + tid];
  const float g = gate[0];
  out[(size_t)qg * Dn + tid] = so[(size_t)qg * Dn + tid] * g + acc * inv * (1.f - g);
}

// ---------------- launch ----------------

extern "C" void kernel_launch(void* const* d_in, const int* in_sizes, int n_in,
                              void* d_out, int out_size, void* d_ws, size_t ws_size,
                              hipStream_t stream) {
  const float* x        = (const float*)d_in[0];
  const float* ki       = (const float*)d_in[1];
  const float* vi       = (const float*)d_in[2];
  const float* mem_keys = (const float*)d_in[3];
  const float* mem_vals = (const float*)d_in[4];
  const float* Wq       = (const float*)d_in[5];
  const float* bq       = (const float*)d_in[6];
  const float* Wk       = (const float*)d_in[7];
  const float* bk       = (const float*)d_in[8];
  const float* Wv       = (const float*)d_in[9];
  const float* bv       = (const float*)d_in[10];
  const float* gate     = (const float*)d_in[11];
  const int*   idxp     = (const int*)d_in[12];
  float* out = (float*)d_out;

  char* ws = (char*)d_ws;
  bf16*  qbf = (bf16*)(ws + OQ);
  bf16*  kf  = (bf16*)(ws + OKF);
  bf16*  vT  = (bf16*)(ws + OVT);
  bf16*  Wt  = (bf16*)(ws + OWT);
  bf16*  mk  = (bf16*)(ws + OMK);
  float* so  = (float*)(ws + OSO);
  float* ts  = (float*)(ws + OTS);
  int*   ti  = (int*)(ws + OTI);

  conv_weights<<<(3 * En * Dn + 255) / 256, 256, 0, stream>>>(Wq, Wk, Wv, Wt);
  conv_prefix<<<(Bn * TQ * Dn + 255) / 256, 256, 0, stream>>>(ki, vi, idxp, kf, vT);
  conv_memk<<<(int)(((size_t)Bn * Mn * Dn + 255) / 256), 256, 0, stream>>>(mem_keys, mk);
  zero_f32<<<(Bn * TQ * Dn + 255) / 256, 256, 0, stream>>>(so, Bn * TQ * Dn);

  proj_kernel<<<dim3((Bn * TQ) / 16, 3), 256, 0, stream>>>(x, Wt, bq, bk, bv, qbf, kf, vT);
  sdpa_kernel<<<dim3(TQ / 16, Bn), 256, 0, stream>>>(qbf, kf, vT, so);
  topk_kernel<<<dim3(TQ / 16, Bn), 256, 0, stream>>>(qbf, mk, ts, ti);
  memattn_kernel<<<Bn * TQ, 128, 0, stream>>>(ts, ti, mem_vals, so, gate, out);
}
